// HybridGraphTransformer_21492016349941
// MI455X (gfx1250) — matrix operand order repaired and checked by reading, not compiled, not run on hardware
//
#include <hip/hip_runtime.h>
#include <hip/hip_bf16.h>
#include <math.h>

typedef __attribute__((ext_vector_type(16))) _Float16 v16h;
typedef __attribute__((ext_vector_type(8)))  float    v8f;

#define WMMA_F16(A, B, C) __builtin_amdgcn_wmma_f32_16x16x32_f16(false, (A), false, (B), (short)0, (C), false, false)

static constexpr int GN   = 2048;   // nodes
static constexpr int GDIN = 128;
static constexpr int GD   = 256;
static constexpr int GL   = 4;
static constexpr int GH   = 8;
static constexpr int GHD  = 32;
static constexpr int GFF  = 1024;
static constexpr int GE   = 65536;

__device__ __forceinline__ float gelu_exact(float x) {
  return 0.5f * x * (1.0f + erff(x * 0.7071067811865475f));
}

// ---------------------------------------------------------------------------
// Generic WMMA GEMM: C[M,N] = act(A[M,K] @ B[K,N] + bias[N] (+ res[M,N]))
// One wave computes one 16x16 tile, K stepped by 32. Row-major f32 in memory,
// converted to f16 fragments in contiguous k order (k = 16*half + s) for A and B.
// ---------------------------------------------------------------------------
template <int ACT, bool RES>
__global__ void hgt_gemm_kernel(const float* __restrict__ A,
                                const float* __restrict__ B,
                                const float* __restrict__ bias,
                                const float* __restrict__ res,
                                float* __restrict__ C,
                                int M, int N, int K) {
  const int lane = threadIdx.x & 31;
  const int wave = threadIdx.x >> 5;
  const int tile = blockIdx.x * 8 + wave;
  const int tilesN = N >> 4;
  const int m0 = (tile / tilesN) << 4;
  const int n0 = (tile % tilesN) << 4;
  if (m0 >= M) return;

  const int half = lane >> 4;   // 0: lanes 0-15, 1: lanes 16-31
  const int l15  = lane & 15;

  v8f acc = {};
  const float* Arow = A + (size_t)(m0 + l15) * K;
  for (int kk = 0; kk < K; kk += 32) {
    v16h af, bf;
    const int ka = kk + half * 16;    // A: lanes<16 hold K {0..15}, lanes>=16 {16..31}
#pragma unroll
    for (int j = 0; j < 8; ++j) {
      af[j]     = (_Float16)Arow[ka + j];
      af[j + 8] = (_Float16)Arow[ka + 8 + j];
    }
    const float* Bp = B + (size_t)(kk + half * 16) * N + n0 + l15;
#pragma unroll
    for (int j = 0; j < 16; ++j) bf[j] = (_Float16)Bp[(size_t)j * N];
    acc = WMMA_F16(af, bf, acc);
  }

  const int col = n0 + l15;
  const float bv = bias[col];
#pragma unroll
  for (int r = 0; r < 8; ++r) {
    const int row = m0 + r + half * 8;   // C/D layout: VGPR r -> rows r / r+8
    float v = acc[r] + bv;
    if (RES) v += res[(size_t)row * N + col];
    if (ACT == 1) v = gelu_exact(v);
    C[(size_t)row * N + col] = v;
  }
}

// ---------------------------------------------------------------------------
// Biased attention, streaming over keys. Block = 8 waves (one per head), one
// 16-query tile per block so all 8 heads of each bias cacheline are consumed
// together. Softmax without max-shift (logits are small; exp cannot overflow
// f32 here): denominator accumulated per-lane, reduced across 16 lanes ONCE
// at the end -> no per-tile ds_bpermute chains, loop is WMMA + bias stream.
// ---------------------------------------------------------------------------
__global__ void hgt_attn_kernel(const float* __restrict__ Q,
                                const float* __restrict__ Kt,
                                const float* __restrict__ V,
                                const float* __restrict__ bias,  // (N,N,H)
                                float* __restrict__ O) {
  __shared__ _Float16 Plds[GH][16][32];
  const int lane = threadIdx.x & 31;
  const int head = threadIdx.x >> 5;
  const int q0   = blockIdx.x * 16;
  const int half = lane >> 4;
  const int l15  = lane & 15;
  const float scale = 0.17677669529663687f;  // 1/sqrt(HD=32)

  // Q fragment (A-layout, 16x32, pre-scaled)
  v16h qf;
  {
    const float* Qr = Q + (size_t)(q0 + l15) * GD + head * GHD;
    const int ka = half * 16;
#pragma unroll
    for (int j = 0; j < 8; ++j) {
      qf[j]     = (_Float16)(Qr[ka + j] * scale);
      qf[j + 8] = (_Float16)(Qr[ka + 8 + j] * scale);
    }
  }

  float ps[8];
  v8f o0 = {}, o1 = {};
#pragma unroll
  for (int r = 0; r < 8; ++r) ps[r] = 0.0f;

  const int nkt = GN / 32;
  for (int kt = 0; kt < nkt; ++kt) {
    const int k0 = kt * 32;

    // Prefetch next key-tile's bias block (16 rows x 32 keys x 8 heads = 128
    // cachelines of 128B); 32 lanes x 4 prefetches covers it exactly.
    if (kt + 1 < nkt) {
#pragma unroll
      for (int j = 0; j < 4; ++j) {
        const int idx = lane + 32 * j;             // 0..127
        const char* p = (const char*)(bias + ((size_t)(q0 + (idx >> 3)) * GN + (k0 + 32)) * GH)
                        + (size_t)(idx & 7) * 128;
        __builtin_prefetch(p, 0, 0);
      }
    }

    // QK^T for two 16-key sub-tiles (K-dim = HD = 32 -> one WMMA each)
    v8f s0 = {}, s1 = {};
    {
      v16h kf;
      const float* Kr0 = Kt + (size_t)(k0 + l15) * GD + head * GHD + half * 16;
#pragma unroll
      for (int j = 0; j < 16; ++j) kf[j] = (_Float16)Kr0[j];
      s0 = WMMA_F16(qf, kf, s0);
      const float* Kr1 = Kt + (size_t)(k0 + 16 + l15) * GD + head * GHD + half * 16;
#pragma unroll
      for (int j = 0; j < 16; ++j) kf[j] = (_Float16)Kr1[j];
      s1 = WMMA_F16(qf, kf, s1);
    }

    // bias add + exp; accumulate denominator per-lane (own columns only)
#pragma unroll
    for (int r = 0; r < 8; ++r) {
      const int q = q0 + r + half * 8;
      const float p0 = __expf(s0[r] + bias[((size_t)q * GN + (k0 + l15)) * GH + head]);
      const float p1 = __expf(s1[r] + bias[((size_t)q * GN + (k0 + 16 + l15)) * GH + head]);
      ps[r] += p0 + p1;
      Plds[head][r + half * 8][l15]      = (_Float16)p0;
      Plds[head][r + half * 8][16 + l15] = (_Float16)p1;
    }
    __syncthreads();
    // Repack P (16x32) into A-fragment layout from LDS
    v16h pf;
    {
      const int klo = half * 16;
#pragma unroll
      for (int j = 0; j < 8; ++j) {
        pf[j]     = Plds[head][l15][klo + j];
        pf[j + 8] = Plds[head][l15][klo + 8 + j];
      }
    }
    __syncthreads();
    // PV: 32 keys x 32 HD via two B-fragments (d-halves)
    {
      v16h vf;
#pragma unroll
      for (int j = 0; j < 16; ++j)
        vf[j] = (_Float16)V[(size_t)(k0 + half * 16 + j) * GD + head * GHD + l15];
      o0 = WMMA_F16(pf, vf, o0);
#pragma unroll
      for (int j = 0; j < 16; ++j)
        vf[j] = (_Float16)V[(size_t)(k0 + half * 16 + j) * GD + head * GHD + 16 + l15];
      o1 = WMMA_F16(pf, vf, o1);
    }
  }

  // one 16-lane reduction per row for the softmax denominator, then store
#pragma unroll
  for (int r = 0; r < 8; ++r) {
    float s = ps[r];
    s += __shfl_xor(s, 1);
    s += __shfl_xor(s, 2);
    s += __shfl_xor(s, 4);
    s += __shfl_xor(s, 8);
    const float inv = 1.0f / s;
    const int q = q0 + r + half * 8;
    O[(size_t)q * GD + head * GHD + l15]      = o0[r] * inv;
    O[(size_t)q * GD + head * GHD + 16 + l15] = o1[r] * inv;
  }
}

// ---------------------------------------------------------------------------
// LayerNorm over D=256, one wave per row (8 rows per 256-thread block)
// ---------------------------------------------------------------------------
__global__ void hgt_ln_kernel(const float* __restrict__ x, const float* __restrict__ g,
                              const float* __restrict__ b, float* __restrict__ y) {
  const int lane = threadIdx.x & 31;
  const int row  = blockIdx.x * 8 + (threadIdx.x >> 5);
  const float* xr = x + (size_t)row * GD;
  float v[8];
  float s = 0.0f;
#pragma unroll
  for (int j = 0; j < 8; ++j) { v[j] = xr[lane * 8 + j]; s += v[j]; }
  for (int mask = 1; mask < 32; mask <<= 1) s += __shfl_xor(s, mask);
  const float mean = s * (1.0f / 256.0f);
  float q = 0.0f;
#pragma unroll
  for (int j = 0; j < 8; ++j) { const float d = v[j] - mean; q += d * d; }
  for (int mask = 1; mask < 32; mask <<= 1) q += __shfl_xor(q, mask);
  const float rstd = rsqrtf(q * (1.0f / 256.0f) + 1e-5f);
  float* yr = y + (size_t)row * GD;
#pragma unroll
  for (int j = 0; j < 8; ++j) {
    const int c = lane * 8 + j;
    yr[c] = (v[j] - mean) * rstd * g[c] + b[c];
  }
}

// edge_w[e] = || edge_attr[e] @ ew_w + ew_b ||
__global__ void hgt_edgew_kernel(const float* __restrict__ ea, const float* __restrict__ w,
                                 const float* __restrict__ b, float* __restrict__ out) {
  const int e = blockIdx.x * blockDim.x + threadIdx.x;
  if (e >= GE) return;
  float a[16];
#pragma unroll
  for (int k = 0; k < 16; ++k) a[k] = ea[(size_t)e * 16 + k];
  float ss = 0.0f;
  for (int c = 0; c < GD; ++c) {
    float acc = b[c];
#pragma unroll
    for (int k = 0; k < 16; ++k) acc += a[k] * w[k * GD + c];
    ss += acc * acc;
  }
  out[e] = sqrtf(ss);
}

__global__ void hgt_fill_kernel(float* __restrict__ p, float v, int n) {
  const int i = blockIdx.x * blockDim.x + threadIdx.x;
  if (i < n) p[i] = v;
}

__global__ void hgt_degree_kernel(const int* __restrict__ dst, float* __restrict__ deg) {
  const int e = blockIdx.x * blockDim.x + threadIdx.x;
  if (e < GE) atomicAdd(&deg[dst[e]], 1.0f);
}

__global__ void hgt_dinv_kernel(float* __restrict__ deg) {
  const int i = blockIdx.x * blockDim.x + threadIdx.x;
  if (i < GN) deg[i] = rsqrtf(deg[i] + 1.0f);
}

// scatter-add: agg[dst] += h[src] * (edge_w * dinv[src]); one block per edge
__global__ void hgt_scatter_kernel(const float* __restrict__ h, const int* __restrict__ src,
                                   const int* __restrict__ dst, const float* __restrict__ ew,
                                   const float* __restrict__ dinv, float* __restrict__ agg) {
  const int e = blockIdx.x;
  const int c = threadIdx.x;
  const int s = src[e], d = dst[e];
  const float coef = ew[e] * dinv[s];
  atomicAdd(&agg[(size_t)d * GD + c], h[(size_t)s * GD + c] * coef);
}

// agg = agg * dinv[row] + h
__global__ void hgt_combine_kernel(float* __restrict__ agg, const float* __restrict__ h,
                                   const float* __restrict__ dinv) {
  const int i = blockIdx.x * blockDim.x + threadIdx.x;
  if (i < GN * GD) agg[i] = agg[i] * dinv[i / GD] + h[i];
}

// gated fusion: g = sigmoid([hl|hg] @ gw + gb); h = g*hl + (1-g)*hg  (wave/row)
__global__ void hgt_gate_kernel(const float* __restrict__ hl, const float* __restrict__ hgm,
                                const float* __restrict__ gw, const float* __restrict__ gb,
                                float* __restrict__ h) {
  const int lane = threadIdx.x & 31;
  const int row  = blockIdx.x * 8 + (threadIdx.x >> 5);
  const float* a  = hl + (size_t)row * GD;
  const float* bb = hgm + (size_t)row * GD;
  float s = 0.0f;
#pragma unroll
  for (int j = 0; j < 8; ++j) {
    const int c = lane * 8 + j;
    s += a[c] * gw[c] + bb[c] * gw[GD + c];
  }
  for (int mask = 1; mask < 32; mask <<= 1) s += __shfl_xor(s, mask);
  const float g = 1.0f / (1.0f + __expf(-(s + gb[0])));
  float* hr = h + (size_t)row * GD;
#pragma unroll
  for (int j = 0; j < 8; ++j) {
    const int c = lane * 8 + j;
    hr[c] = g * a[c] + (1.0f - g) * bb[c];
  }
}

__global__ void hgt_mean_kernel(const float* __restrict__ h, float* __restrict__ out) {
  const int c = threadIdx.x;  // 256 threads
  float s = 0.0f;
  for (int r = 0; r < GN; ++r) s += h[(size_t)r * GD + c];
  out[c] = s * (1.0f / (float)GN);
}

// ---------------------------------------------------------------------------
extern "C" void kernel_launch(void* const* d_in, const int* in_sizes, int n_in,
                              void* d_out, int out_size, void* d_ws, size_t ws_size,
                              hipStream_t stream) {
  (void)in_sizes; (void)n_in; (void)out_size; (void)ws_size;
  const float* x     = (const float*)d_in[0];
  const int*   ei    = (const int*)d_in[1];       // (2,E)
  const float* ea    = (const float*)d_in[2];
  const float* abias = (const float*)d_in[3];     // (N,N,H)
  const float* in_w  = (const float*)d_in[4];
  const float* in_b  = (const float*)d_in[5];
  const float* ew_w  = (const float*)d_in[6];
  const float* ew_b  = (const float*)d_in[7];
  const float* LW  = (const float*)d_in[8];
  const float* LB  = (const float*)d_in[9];
  const float* LNG = (const float*)d_in[10];
  const float* LNB = (const float*)d_in[11];
  const float* QW  = (const float*)d_in[12];
  const float* QB  = (const float*)d_in[13];
  const float* KW  = (const float*)d_in[14];
  const float* KB  = (const float*)d_in[15];
  const float* VW  = (const float*)d_in[16];
  const float* VB  = (const float*)d_in[17];
  const float* OW  = (const float*)d_in[18];
  const float* OB  = (const float*)d_in[19];
  const float* F1W = (const float*)d_in[20];
  const float* F1B = (const float*)d_in[21];
  const float* F2W = (const float*)d_in[22];
  const float* F2B = (const float*)d_in[23];
  const float* N1G = (const float*)d_in[24];
  const float* N1B = (const float*)d_in[25];
  const float* N2G = (const float*)d_in[26];
  const float* N2B = (const float*)d_in[27];
  const float* GW  = (const float*)d_in[28];
  const float* GB  = (const float*)d_in[29];
  const float* FNG = (const float*)d_in[30];
  const float* FNB = (const float*)d_in[31];
  float* out = (float*)d_out;

  const int* src = ei;
  const int* dst = ei + GE;

  float* ws = (float*)d_ws;
  size_t off = 0;
  auto alloc = [&](size_t n) { float* p = ws + off; off += n; return p; };
  const size_t ND = (size_t)GN * GD;
  float* h    = alloc(ND);
  float* hn   = alloc(ND);
  float* Qb   = alloc(ND);
  float* Kb   = alloc(ND);
  float* Vb   = alloc(ND);
  float* Ob   = alloc(ND);
  float* hg   = alloc(ND);
  float* hl   = alloc(ND);
  float* agg  = alloc(ND);
  float* tmpB = alloc(ND);
  float* ln2  = alloc(ND);
  float* hgm  = alloc(ND);
  float* ffb  = alloc((size_t)GN * GFF);
  float* ew   = alloc(GE);
  float* dinv = alloc(GN);

  const dim3 B256(256);
  const int tileBlk_D  = (GN / 16) * (GD / 16) / 8;   // 256 blocks for N=256 GEMMs
  const int tileBlk_FF = (GN / 16) * (GFF / 16) / 8;  // 1024 blocks for N=1024 GEMM

  // edge weights + degree normalization
  hgt_edgew_kernel<<<GE / 256, B256, 0, stream>>>(ea, ew_w, ew_b, ew);
  hgt_fill_kernel<<<(GN + 255) / 256, B256, 0, stream>>>(dinv, 0.0f, GN);
  hgt_degree_kernel<<<GE / 256, B256, 0, stream>>>(dst, dinv);
  hgt_dinv_kernel<<<(GN + 255) / 256, B256, 0, stream>>>(dinv);

  // input projection: h = x @ in_w + in_b   (K = 128)
  hgt_gemm_kernel<0, false><<<tileBlk_D, B256, 0, stream>>>(x, in_w, in_b, nullptr, h, GN, GD, GDIN);

  for (int i = 0; i < GL; ++i) {
    const float* lw  = LW + (size_t)i * GD * GD;
    const float* lb  = LB + (size_t)i * GD;
    const float* lng = LNG + (size_t)i * GD;
    const float* lnb = LNB + (size_t)i * GD;
    const float* qw  = QW + (size_t)i * GD * GD;
    const float* qb  = QB + (size_t)i * GD;
    const float* kw  = KW + (size_t)i * GD * GD;
    const float* kb  = KB + (size_t)i * GD;
    const float* vw  = VW + (size_t)i * GD * GD;
    const float* vb  = VB + (size_t)i * GD;
    const float* ow  = OW + (size_t)i * GD * GD;
    const float* ob  = OB + (size_t)i * GD;
    const float* f1w = F1W + (size_t)i * GD * GFF;
    const float* f1b = F1B + (size_t)i * GFF;
    const float* f2w = F2W + (size_t)i * GFF * GD;
    const float* f2b = F2B + (size_t)i * GD;
    const float* n1g = N1G + (size_t)i * GD;
    const float* n1b = N1B + (size_t)i * GD;
    const float* n2g = N2G + (size_t)i * GD;
    const float* n2b = N2B + (size_t)i * GD;
    const float* gw  = GW + (size_t)i * 2 * GD;
    const float* gb  = GB + (size_t)i;

    // ---- local GCN ----
    hgt_fill_kernel<<<(int)(ND / 256), B256, 0, stream>>>(agg, 0.0f, (int)ND);
    hgt_scatter_kernel<<<GE, B256, 0, stream>>>(h, src, dst, ew, dinv, agg);
    hgt_combine_kernel<<<(int)(ND / 256), B256, 0, stream>>>(agg, h, dinv);
    hgt_gemm_kernel<1, false><<<tileBlk_D, B256, 0, stream>>>(agg, lw, lb, nullptr, tmpB, GN, GD, GD);
    hgt_ln_kernel<<<GN / 8, B256, 0, stream>>>(tmpB, lng, lnb, hl);

    // ---- global biased attention ----
    hgt_ln_kernel<<<GN / 8, B256, 0, stream>>>(h, n1g, n1b, hn);
    hgt_gemm_kernel<0, false><<<tileBlk_D, B256, 0, stream>>>(hn, qw, qb, nullptr, Qb, GN, GD, GD);
    hgt_gemm_kernel<0, false><<<tileBlk_D, B256, 0, stream>>>(hn, kw, kb, nullptr, Kb, GN, GD, GD);
    hgt_gemm_kernel<0, false><<<tileBlk_D, B256, 0, stream>>>(hn, vw, vb, nullptr, Vb, GN, GD, GD);
    hgt_attn_kernel<<<GN / 16, B256, 0, stream>>>(Qb, Kb, Vb, abias, Ob);
    hgt_gemm_kernel<0, true><<<tileBlk_D, B256, 0, stream>>>(Ob, ow, ob, h, hg, GN, GD, GD);
    hgt_ln_kernel<<<GN / 8, B256, 0, stream>>>(hg, n2g, n2b, ln2);
    hgt_gemm_kernel<1, false><<<tileBlk_FF, B256, 0, stream>>>(ln2, f1w, f1b, nullptr, ffb, GN, GFF, GD);
    hgt_gemm_kernel<0, true><<<tileBlk_D, B256, 0, stream>>>(ffb, f2w, f2b, hg, hgm, GN, GD, GFF);

    // ---- gated fusion ----
    hgt_gate_kernel<<<GN / 8, B256, 0, stream>>>(hl, hgm, gw, gb, h);
  }

  // final LN into d_out, then column mean
  hgt_ln_kernel<<<GN / 8, B256, 0, stream>>>(h, FNG, FNB, out);
  hgt_mean_kernel<<<1, B256, 0, stream>>>(out, out + ND);
}
